// LiquidNeuralNetwork_49701361549748
// MI455X (gfx1250) — matrix operand order repaired and checked by reading, not compiled, not run on hardware
//
#include <hip/hip_runtime.h>
#include <math.h>

// ---------------- problem constants ----------------
#define BATCH 64
#define SEQ   256
#define INSZ  512
#define HID   1024
#define INV_TAU 2.0f          // 1/0.5

// ---------------- launch shape ----------------
#define NBLK  32              // 32 workgroups, all co-resident (persistent kernel)
#define NTHR  256             // 8 wave32 per workgroup
#define NWAVE 8

typedef _Float16 v8h  __attribute__((ext_vector_type(8)));
typedef _Float16 v16h __attribute__((ext_vector_type(16)));
typedef float    v8f  __attribute__((ext_vector_type(8)));

// Force global (addrspace(1)) accesses in the hot loops: global and generic
// share address values for global memory on amdgcn, so the int round-trip is a
// no-op address-wise but pins the backend to global_load/global_store (flat
// loads would burn DScnt + the LDS path per the CDNA5 ISA).
#define GLD_V8H(p)   (*(const __attribute__((address_space(1))) v8h*)(uintptr_t)(p))
#define GLD_F32(p)   (*(const __attribute__((address_space(1))) float*)(uintptr_t)(p))
#define GST_F32(p,v) (*(__attribute__((address_space(1))) float*)(uintptr_t)(p) = (v))
#define GST_F16(p,v) (*(__attribute__((address_space(1))) _Float16*)(uintptr_t)(p) = (v))

// ---------------- workspace layout (bytes) ----------------
// [0,256)   grid-barrier counter (zeroed by hipMemsetAsync each launch)
// then f16 copies of x / W0 / U0 / W1 / U1, double-buffered f16 h, f32 h.
static constexpr size_t OFF_BAR  = 0;
static constexpr size_t OFF_XF   = 256;
static constexpr size_t SZ_XF    = (size_t)BATCH * SEQ * INSZ * 2;
static constexpr size_t SZ_H16   = (size_t)BATCH * HID * 2;
static constexpr size_t OFF_W0   = OFF_XF + SZ_XF;
static constexpr size_t OFF_U0   = OFF_W0 + (size_t)HID * INSZ * 2;
static constexpr size_t OFF_W1   = OFF_U0 + (size_t)HID * HID * 2;
static constexpr size_t OFF_U1   = OFF_W1 + (size_t)HID * HID * 2;
static constexpr size_t OFF_H0   = OFF_U1 + (size_t)HID * HID * 2;   // h0 f16 buf 0,1
static constexpr size_t OFF_H1   = OFF_H0 + 2 * SZ_H16;              // h1 f16 buf 0,1
static constexpr size_t OFF_H032 = OFF_H1 + 2 * SZ_H16;              // h0 f32 (owner-private)
static constexpr size_t OFF_H132 = OFF_H032 + (size_t)BATCH * HID * 4;

// ---------------- grid-wide barrier (atomic, monotonic target) ----------------
__device__ __forceinline__ void grid_barrier(unsigned* bar, unsigned target) {
    __threadfence();                 // make this thread's global stores visible device-wide
    __syncthreads();
    if (threadIdx.x == 0) {
        __hip_atomic_fetch_add(bar, 1u, __ATOMIC_ACQ_REL, __HIP_MEMORY_SCOPE_AGENT);
        while (__hip_atomic_load(bar, __ATOMIC_ACQUIRE, __HIP_MEMORY_SCOPE_AGENT) < target) {
            __builtin_amdgcn_s_sleep(1);
        }
    }
    __syncthreads();
    __threadfence();                 // acquire side for the other 255 threads
}

// ---------------- WMMA fragment loaders ----------------
// A operand: 16x32 f16 (MxK). ISA layout: lanes 0-15 hold row M=lane with K
// {k0..k0+7, k0+16..k0+23}; lanes 16-31 same rows, K shifted by 8.
__device__ __forceinline__ v16h load_a_frag(const _Float16* __restrict__ base,
                                            int ld, int m0, int k0, int lane) {
    const int half = lane >> 4;                         // 0 or 1
    const _Float16* p = base + (size_t)(m0 + (lane & 15)) * ld + k0 + half * 8;
    v8h lo = GLD_V8H(p);                                // 16B aligned
    v8h hi = GLD_V8H(p + 16);
    return __builtin_shufflevector(lo, hi, 0,1,2,3,4,5,6,7,8,9,10,11,12,13,14,15);
}

// B operand: 32x16 f16 (KxN), row striped across lanes. Lane holds column
// n = n0+(lane&15), K = {k0..k0+7, k0+16..k0+23} (lanes 0-15), K shifted by 8 (16-31).
// Original weight layout (N rows, K cols, row-major) gives K-contiguous loads.
__device__ __forceinline__ v16h load_b_frag(const _Float16* __restrict__ wgt,
                                            int ld, int n0, int k0, int lane) {
    const int half = lane >> 4;
    const _Float16* p = wgt + (size_t)(n0 + (lane & 15)) * ld + k0 + half * 8;
    v8h lo = GLD_V8H(p);
    v8h hi = GLD_V8H(p + 16);
    return __builtin_shufflevector(lo, hi, 0,1,2,3,4,5,6,7,8,9,10,11,12,13,14,15);
}

// acc += A(MxK) * W^T  via v_wmma_f32_16x16x32_f16 along K
__device__ __forceinline__ v8f gemm_acc(v8f acc,
                                        const _Float16* __restrict__ A, int lda,
                                        const _Float16* __restrict__ W, int ldw,
                                        int K, int m0, int n0, int lane) {
#pragma unroll 8
    for (int k0 = 0; k0 < K; k0 += 32) {
        v16h a = load_a_frag(A, lda, m0, k0, lane);
        v16h b = load_b_frag(W, ldw, n0, k0, lane);
        acc = __builtin_amdgcn_wmma_f32_16x16x32_f16(
            /*neg_a=*/false, a, /*neg_b=*/false, b,
            /*c_mod=*/(short)0, acc, /*reuse_a=*/false, /*reuse_b=*/false);
    }
    return acc;
}

// C/D layout: VGPR r, lanes 0-15 -> (m0+r, n0+lane); lanes 16-31 -> (m0+8+r, n0+lane-16)
__device__ __forceinline__ void epilogue(v8f acc,
                                         const float* __restrict__ bias,
                                         const float* __restrict__ alpha,
                                         float* __restrict__ h32,
                                         _Float16* __restrict__ h16w,
                                         int m0, int n0, int lane,
                                         float* __restrict__ out_or_null) {
    const int n = n0 + (lane & 15);
    const int mbase = m0 + ((lane >> 4) << 3);
    const float bn = GLD_F32(bias + n);
    const float an = GLD_F32(alpha + n) * INV_TAU;
#pragma unroll
    for (int r = 0; r < 8; ++r) {
        const size_t idx = (size_t)(mbase + r) * HID + n;
        const float hold = GLD_F32(h32 + idx);
        const float pre  = tanhf(acc[r] + bn);
        const float hnew = hold + an * (pre - hold);
        GST_F32(h32 + idx, hnew);
        GST_F16(h16w + idx, (_Float16)hnew);
        if (out_or_null) GST_F32(out_or_null + idx, hnew);
    }
}

// ---------------- persistent kernel ----------------
__global__ __launch_bounds__(NTHR, 1)
void lnn_persistent(const float* __restrict__ x,
                    const float* __restrict__ W0, const float* __restrict__ U0,
                    const float* __restrict__ b0, const float* __restrict__ a0,
                    const float* __restrict__ W1, const float* __restrict__ U1,
                    const float* __restrict__ b1, const float* __restrict__ a1,
                    float* __restrict__ out, char* __restrict__ ws) {
    unsigned* bar = (unsigned*)(ws + OFF_BAR);
    _Float16* xf  = (_Float16*)(ws + OFF_XF);
    _Float16* w0f = (_Float16*)(ws + OFF_W0);
    _Float16* u0f = (_Float16*)(ws + OFF_U0);
    _Float16* w1f = (_Float16*)(ws + OFF_W1);
    _Float16* u1f = (_Float16*)(ws + OFF_U1);
    float* h032 = (float*)(ws + OFF_H032);
    float* h132 = (float*)(ws + OFF_H132);

    const int lane = threadIdx.x & 31;
    const int wave = threadIdx.x >> 5;
    // 256 waves <-> 256 output tiles: 4 M-tiles (batch) x 64 N-tiles (hidden)
    const int tile = blockIdx.x * NWAVE + wave;
    const int m0 = (tile & 3) << 4;
    const int n0 = (tile >> 2) << 4;

    // ---- phase 0: f32 -> f16 conversion of weights and x; zero hidden state ----
    {
        const size_t gtid = (size_t)blockIdx.x * NTHR + threadIdx.x;
        const size_t gs = (size_t)NBLK * NTHR;
        for (size_t i = gtid; i < (size_t)HID * INSZ; i += gs) w0f[i] = (_Float16)W0[i];
        for (size_t i = gtid; i < (size_t)HID * HID;  i += gs) u0f[i] = (_Float16)U0[i];
        for (size_t i = gtid; i < (size_t)HID * HID;  i += gs) w1f[i] = (_Float16)W1[i];
        for (size_t i = gtid; i < (size_t)HID * HID;  i += gs) u1f[i] = (_Float16)U1[i];
        for (size_t i = gtid; i < (size_t)BATCH * SEQ * INSZ; i += gs) xf[i] = (_Float16)x[i];
        _Float16* h0b = (_Float16*)(ws + OFF_H0);
        _Float16* h1b = (_Float16*)(ws + OFF_H1);
        for (size_t i = gtid; i < (size_t)BATCH * HID; i += gs) {
            h032[i] = 0.0f; h132[i] = 0.0f;
            h0b[i] = (_Float16)0.0f; h0b[i + BATCH * HID] = (_Float16)0.0f;
            h1b[i] = (_Float16)0.0f; h1b[i + BATCH * HID] = (_Float16)0.0f;
        }
    }
    unsigned bc = 1;
    grid_barrier(bar, bc * NBLK);

    // ---- phase 1: 256 sequential Euler steps, double-buffered f16 h ----
    // Double-buffer selection is scalar offset arithmetic on one base pointer
    // (NOT a pointer array) so infer-address-space keeps everything global.
    const v8f vzero = {};
    for (int t = 0; t < SEQ; ++t) {
        const size_t cur = (size_t)(t & 1) * SZ_H16;
        const size_t nxt = (size_t)((t + 1) & 1) * SZ_H16;
        const _Float16* h0_cur = (const _Float16*)(ws + OFF_H0 + cur);
        _Float16*       h0_nxt = (_Float16*)(ws + OFF_H0 + nxt);
        const _Float16* h1_cur = (const _Float16*)(ws + OFF_H1 + cur);
        _Float16*       h1_nxt = (_Float16*)(ws + OFF_H1 + nxt);

        // prefetch next step's x rows into cache hierarchy (global_prefetch_b8)
        if (t + 1 < SEQ) {
            const _Float16* pf = xf + (size_t)(t + 1) * INSZ
                               + (size_t)(m0 + (lane & 15)) * (SEQ * INSZ);
            __builtin_prefetch((const void*)pf, 0, 1);
        }

        // layer 0: h0' from tanh(x_t W0^T + h0 U0^T + b0)
        v8f acc = vzero;
        acc = gemm_acc(acc, xf + (size_t)t * INSZ, SEQ * INSZ, w0f, INSZ, INSZ, m0, n0, lane);
        acc = gemm_acc(acc, h0_cur, HID, u0f, HID, HID, m0, n0, lane);
        epilogue(acc, b0, a0, h032, h0_nxt, m0, n0, lane, nullptr);
        ++bc; grid_barrier(bar, bc * NBLK);

        // layer 1: h1' from tanh(h0' W1^T + h1 U1^T + b1)
        acc = vzero;
        acc = gemm_acc(acc, h0_nxt, HID, w1f, HID, HID, m0, n0, lane);
        acc = gemm_acc(acc, h1_cur, HID, u1f, HID, HID, m0, n0, lane);
        epilogue(acc, b1, a1, h132, h1_nxt, m0, n0, lane,
                 (t == SEQ - 1) ? out : nullptr);
        ++bc; grid_barrier(bar, bc * NBLK);
    }
}

extern "C" void kernel_launch(void* const* d_in, const int* in_sizes, int n_in,
                              void* d_out, int out_size, void* d_ws, size_t ws_size,
                              hipStream_t stream) {
    const float* x  = (const float*)d_in[0];
    const float* W0 = (const float*)d_in[1];
    const float* U0 = (const float*)d_in[2];
    const float* b0 = (const float*)d_in[3];
    const float* a0 = (const float*)d_in[4];
    const float* W1 = (const float*)d_in[5];
    const float* U1 = (const float*)d_in[6];
    const float* b1 = (const float*)d_in[7];
    const float* a1 = (const float*)d_in[8];
    float* out = (float*)d_out;
    char* ws = (char*)d_ws;

    // zero the grid-barrier counter every launch (graph-capture safe)
    hipMemsetAsync(ws + OFF_BAR, 0, 256, stream);
    lnn_persistent<<<NBLK, NTHR, 0, stream>>>(x, W0, U0, b0, a0, W1, U1, b1, a1, out, ws);
}
